// ChildSumTreeLSTM_22385369547044
// MI455X (gfx1250) — matrix operand order repaired and checked by reading, not compiled, not run on hardware
//
#include <hip/hip_runtime.h>
#include <hip/hip_bf16.h>
#include <stdint.h>

// ---------------------------------------------------------------------------
// Child-Sum TreeLSTM for MI455X (gfx1250, wave32, WMMA bf16 16x16x32).
//
//   Wx = concat(Wioux[768,256], Wfx[256,256])  -> [1024,256]  (bf16, B-layout)
//   Wh = concat(Wiouh[768,256], Wfh[256,256])  -> [1024,256]  (bf16, B-layout)
//   bcat = concat(bioux, bfx)                  -> [1024]
// Per level (32 parents / block, 2 M-tiles, B-frags held in regs across both):
//   iou  = x@Wx_iou + (hL+hR)@Wh_iou + b ;  f_k = sig(x@Wx_f + h_k@Wh_f + b + fb)
//   c    = sig(i)*tanh(u) + f_l*c_l + f_r*c_r ;  h = sig(o)*tanh(c)
// bf16 copies of x and h live in workspace so LDS staging is pure data
// movement via global_load_async_to_lds_b128 (ASYNCcnt) on full tiles.
// ---------------------------------------------------------------------------

typedef unsigned short u16;
typedef __attribute__((ext_vector_type(8)))  float   v8f;
typedef __attribute__((ext_vector_type(8)))  u16     v8u;
typedef __attribute__((ext_vector_type(16))) u16     v16u;
typedef __attribute__((ext_vector_type(16))) __bf16  v16bf;

#define IN_DIM   256
#define MEM_DIM  256
#define PROJ     1024   // 768 (iou) + 256 (f)
#define KSTEPS   8      // 256 / 32
#define MTILES   2      // 32 parents per block

__device__ __forceinline__ u16 f2bf(float f) {
  union { float f; uint32_t u; } v; v.f = f;
  uint32_t u = v.u;
  uint32_t r = u + 0x7FFFu + ((u >> 16) & 1u);   // round-to-nearest-even
  return (u16)(r >> 16);
}

__device__ __forceinline__ uint32_t pack2bf(float a, float b) {
  return (uint32_t)f2bf(a) | ((uint32_t)f2bf(b) << 16);
}

__device__ __forceinline__ float sigm(float x) {
  return 1.0f / (1.0f + __expf(-x));
}

__device__ __forceinline__ v8f wmma_bf16(v16u a, v16u b, v8f c) {
  return __builtin_amdgcn_wmma_f32_16x16x32_bf16(
      false, __builtin_bit_cast(v16bf, a),
      false, __builtin_bit_cast(v16bf, b),
      (short)0, c, false, false);
}

// Async copy 16B global -> LDS (per-lane). VDST VGPR holds the LDS byte
// offset (low 32 bits of the flat LDS address), ADDR holds the 64-bit
// global address. Tracked by ASYNCcnt.
__device__ __forceinline__ void async_ld16(u16* lds_dst, const u16* g_src) {
  asm volatile("global_load_async_to_lds_b128 %0, %1, off"
               :: "v"((uint32_t)(uintptr_t)lds_dst), "v"(g_src)
               : "memory");
}
__device__ __forceinline__ void wait_async0() {
  asm volatile("s_wait_asynccnt 0" ::: "memory");
}

// A-fragment (16x32 bf16) from an LDS tile stored row-major [16][256] bf16.
// lane l: row m=l%16; K chunks [8h,8h+8) and [16+8h,16+8h+8), h=l/16.
__device__ __forceinline__ v16u load_a(const u16* tile, int lane, int ks) {
  int m = lane & 15, half = lane >> 4;
  const v8u* row = (const v8u*)(tile + m * 256);
  v8u lo = row[ks * 4 + half];
  v8u hi = row[ks * 4 + half + 2];
  return __builtin_shufflevector(lo, hi, 0, 1, 2, 3, 4, 5, 6, 7,
                                         8, 9, 10, 11, 12, 13, 14, 15);
}

// B-fragment (32x16 bf16) pre-packed: fragment (t, ks) = 32 lanes x 16 bf16,
// lane = (khalf<<4)|ncol.
__device__ __forceinline__ v16u load_b(const u16* W, int t, int ks, int lane) {
  const v16u* g = (const v16u*)W;
  return g[(t * KSTEPS + ks) * 32 + lane];
}

// ---------------------------------------------------------------------------
// Prep 1: pack fp32 weights into bf16 WMMA-B layout + concat biases.
// ---------------------------------------------------------------------------
__global__ __launch_bounds__(256) void treelstm_pack_kernel(
    const float* __restrict__ Wioux, const float* __restrict__ Wfx,
    const float* __restrict__ Wiouh, const float* __restrict__ Wfh,
    const float* __restrict__ bioux, const float* __restrict__ bfx,
    u16* __restrict__ WxB, u16* __restrict__ WhB, float* __restrict__ bcat) {
  int id = blockIdx.x * 256 + threadIdx.x;      // 0 .. 1024*256-1
  int n = id >> 8;
  int k = id & 255;

  float wx = (n < 768) ? Wioux[n * IN_DIM + k] : Wfx[(n - 768) * IN_DIM + k];
  float wh = (n < 768) ? Wiouh[n * MEM_DIM + k] : Wfh[(n - 768) * MEM_DIM + k];

  int t = n >> 4, nl = n & 15;
  int s = k >> 5, kk = k & 31;
  int half = (kk >> 3) & 1, ke = (kk & 7) + ((kk >> 4) << 3);
  size_t idx = ((size_t)(t * KSTEPS + s) * 32 + half * 16 + nl) * 16 + ke;
  WxB[idx] = f2bf(wx);
  WhB[idx] = f2bf(wh);

  if (id < PROJ) bcat[id] = (id < 768) ? bioux[id] : bfx[id - 768];
}

// ---------------------------------------------------------------------------
// Prep 2: inputs fp32 -> bf16 mirror (one bandwidth pass).
// ---------------------------------------------------------------------------
__global__ __launch_bounds__(256) void treelstm_xbf_kernel(
    const float* __restrict__ x, u16* __restrict__ xbf, long total) {
  long i = ((long)blockIdx.x * 256 + threadIdx.x) * 4;
  if (i < total) {                      // total % 4 == 0
    float4 v = *(const float4*)(x + i);
    uint2 o;
    o.x = pack2bf(v.x, v.y);
    o.y = pack2bf(v.z, v.w);
    *(uint2*)(xbf + i) = o;
  }
}

// ---------------------------------------------------------------------------
// Fused per-level kernel: 32 parents / block (2 M-tiles), 8 waves, each wave
// owns a 32-column slice of the memory dim (2 N-tiles of 16 per gate).
// K-step outer: 8 B-fragments loaded once, reused by both M-tiles.
// ---------------------------------------------------------------------------
__global__ __launch_bounds__(256) void treelstm_level_kernel(
    const u16* __restrict__ xbf, const u16* __restrict__ hbf,
    const u16* __restrict__ WxB, const u16* __restrict__ WhB,
    const float* __restrict__ bcat, const float* __restrict__ fb,
    float* __restrict__ h_all, float* __restrict__ c_all,
    int s_node, int e_node, int has_children) {
  __shared__ u16 aX[32 * 256];
  __shared__ u16 aL[32 * 256];
  __shared__ u16 aR[32 * 256];

  const int tid = threadIdx.x;
  const int tile_base = s_node + blockIdx.x * 32;

  // ---- stage A tiles (bf16) into LDS -------------------------------------
  {
    int row = tid >> 3;                 // 0..31
    int c0 = (tid & 7) * 32;            // 32 cols = 4 x 16B per matrix
    int p = tile_base + row;
    bool full_tile = (tile_base + 32) <= e_node;   // block-uniform
    if (full_tile) {
      const u16* gx = xbf + (size_t)p * IN_DIM + c0;
      u16* lx = &aX[row * 256 + c0];
#pragma unroll
      for (int q = 0; q < 4; ++q) async_ld16(lx + q * 8, gx + q * 8);
      if (has_children) {
        size_t cl = 2 * (size_t)p + 1;
        const u16* gl = hbf + cl * MEM_DIM + c0;
        const u16* gr = hbf + (cl + 1) * MEM_DIM + c0;
        u16* ll = &aL[row * 256 + c0];
        u16* lr = &aR[row * 256 + c0];
#pragma unroll
        for (int q = 0; q < 4; ++q) {
          async_ld16(ll + q * 8, gl + q * 8);
          async_ld16(lr + q * 8, gr + q * 8);
        }
      }
      wait_async0();
    } else {
      // partial tile (only the tiny top levels): padded copy through VGPRs
      bool valid = p < e_node;
      v8u zero = {0, 0, 0, 0, 0, 0, 0, 0};
      const v8u* gx = (const v8u*)(xbf + (size_t)p * IN_DIM + c0);
      v8u* lx = (v8u*)&aX[row * 256 + c0];
#pragma unroll
      for (int q = 0; q < 4; ++q) lx[q] = valid ? gx[q] : zero;
      if (has_children) {
        size_t cl = 2 * (size_t)p + 1;
        const v8u* gl = (const v8u*)(hbf + cl * MEM_DIM + c0);
        const v8u* gr = (const v8u*)(hbf + (cl + 1) * MEM_DIM + c0);
        v8u* ll = (v8u*)&aL[row * 256 + c0];
        v8u* lr = (v8u*)&aR[row * 256 + c0];
#pragma unroll
        for (int q = 0; q < 4; ++q) {
          ll[q] = valid ? gl[q] : zero;
          lr[q] = valid ? gr[q] : zero;
        }
      }
    }
  }
  __syncthreads();

  const int lane = tid & 31;
  const int wave = tid >> 5;

  for (int sub = 0; sub < 2; ++sub) {
    const int jt = wave * 2 + sub;      // N-tile within the 256-wide mem dim
    v8f ai[MTILES], ao[MTILES], au[MTILES], afl[MTILES], afr[MTILES];
#pragma unroll
    for (int mt = 0; mt < MTILES; ++mt) {
      v8f z = {0, 0, 0, 0, 0, 0, 0, 0};
      ai[mt] = z; ao[mt] = z; au[mt] = z; afl[mt] = z; afr[mt] = z;
    }

#pragma unroll
    for (int ks = 0; ks < KSTEPS; ++ks) {
      // B fragments: loaded once per ks, reused by both M-tiles.
      v16u bxi = load_b(WxB, jt, ks, lane);
      v16u bxo = load_b(WxB, 16 + jt, ks, lane);
      v16u bxu = load_b(WxB, 32 + jt, ks, lane);
      // prefetch next ks fragments into cache
      __builtin_prefetch((const v16u*)WxB + (jt * KSTEPS + ((ks + 1) & 7)) * 32 + lane, 0, 3);
      if (has_children) {
        v16u bhi = load_b(WhB, jt, ks, lane);
        v16u bho = load_b(WhB, 16 + jt, ks, lane);
        v16u bhu = load_b(WhB, 32 + jt, ks, lane);
        v16u bxf = load_b(WxB, 48 + jt, ks, lane);
        v16u bhf = load_b(WhB, 48 + jt, ks, lane);
        __builtin_prefetch((const v16u*)WhB + (jt * KSTEPS + ((ks + 1) & 7)) * 32 + lane, 0, 3);
#pragma unroll
        for (int mt = 0; mt < MTILES; ++mt) {
          const u16* bX = aX + mt * 16 * 256;
          const u16* bL = aL + mt * 16 * 256;
          const u16* bR = aR + mt * 16 * 256;
          v16u ax = load_a(bX, lane, ks);
          v16u al = load_a(bL, lane, ks);
          v16u ar = load_a(bR, lane, ks);
          ai[mt] = wmma_bf16(ax, bxi, ai[mt]);
          ai[mt] = wmma_bf16(al, bhi, ai[mt]);
          ai[mt] = wmma_bf16(ar, bhi, ai[mt]);
          ao[mt] = wmma_bf16(ax, bxo, ao[mt]);
          ao[mt] = wmma_bf16(al, bho, ao[mt]);
          ao[mt] = wmma_bf16(ar, bho, ao[mt]);
          au[mt] = wmma_bf16(ax, bxu, au[mt]);
          au[mt] = wmma_bf16(al, bhu, au[mt]);
          au[mt] = wmma_bf16(ar, bhu, au[mt]);
          afl[mt] = wmma_bf16(ax, bxf, afl[mt]);
          afl[mt] = wmma_bf16(al, bhf, afl[mt]);
          afr[mt] = wmma_bf16(ax, bxf, afr[mt]);
          afr[mt] = wmma_bf16(ar, bhf, afr[mt]);
        }
      } else {
#pragma unroll
        for (int mt = 0; mt < MTILES; ++mt) {
          v16u ax = load_a(aX + mt * 16 * 256, lane, ks);
          ai[mt] = wmma_bf16(ax, bxi, ai[mt]);
          ao[mt] = wmma_bf16(ax, bxo, ao[mt]);
          au[mt] = wmma_bf16(ax, bxu, au[mt]);
        }
      }
    }

    // ---- gate epilogue (C layout: n = lane%16, m = vgpr + 8*(lane/16)) ----
    const int n = lane & 15;
    const int hset = lane >> 4;
    const int j = jt * 16 + n;
    const float bi = bcat[j];
    const float bo = bcat[256 + j];
    const float bu = bcat[512 + j];
    const float bfj = bcat[768 + j] + fb[j];
#pragma unroll
    for (int mt = 0; mt < MTILES; ++mt) {
#pragma unroll
      for (int r = 0; r < 8; ++r) {
        int m = mt * 16 + r + 8 * hset;
        int p = tile_base + m;
        if (p < e_node) {
          float iv = sigm(ai[mt][r] + bi);
          float ov = sigm(ao[mt][r] + bo);
          float uv = tanhf(au[mt][r] + bu);
          float c = iv * uv;
          if (has_children) {
            size_t cl = 2 * (size_t)p + 1;
            float flv = sigm(afl[mt][r] + bfj);
            float frv = sigm(afr[mt][r] + bfj);
            c += flv * c_all[cl * MEM_DIM + j] +
                 frv * c_all[(cl + 1) * MEM_DIM + j];
          }
          float h = ov * tanhf(c);
          size_t o = (size_t)p * MEM_DIM + j;
          c_all[o] = c;
          h_all[o] = h;
          ((u16*)hbf)[o] = f2bf(h);    // bf16 mirror for the parent level
        }
      }
    }
  }
}

// Copy root h/c to the first two output slots.
__global__ void treelstm_root_kernel(float* __restrict__ out,
                                     const float* __restrict__ h_all,
                                     const float* __restrict__ c_all) {
  int t = threadIdx.x;
  if (t < MEM_DIM) {
    out[t] = h_all[t];
    out[MEM_DIM + t] = c_all[t];
  }
}

extern "C" void kernel_launch(void* const* d_in, const int* in_sizes, int n_in,
                              void* d_out, int out_size, void* d_ws, size_t ws_size,
                              hipStream_t stream) {
  const float* inputs = (const float*)d_in[0];
  const float* Wioux  = (const float*)d_in[1];
  const float* bioux  = (const float*)d_in[2];
  const float* Wiouh  = (const float*)d_in[3];
  const float* Wfx    = (const float*)d_in[4];
  const float* bfx    = (const float*)d_in[5];
  const float* Wfh    = (const float*)d_in[6];
  const float* fb     = (const float*)d_in[7];

  const int N = in_sizes[0] / IN_DIM;          // 2^depth - 1
  int depth = 0;
  for (int t = N + 1; t > 1; t >>= 1) depth++; // 18

  float* out   = (float*)d_out;
  float* h_all = out + 2 * MEM_DIM;            // h_all lives inside d_out

  char* ws = (char*)d_ws;
  float* c_all = (float*)ws;
  size_t off = (size_t)N * MEM_DIM * sizeof(float);
  u16* xbf = (u16*)(ws + off);   off += (size_t)N * IN_DIM * sizeof(u16);
  u16* hbf = (u16*)(ws + off);   off += (size_t)N * MEM_DIM * sizeof(u16);
  u16* WxB = (u16*)(ws + off);   off += (size_t)PROJ * IN_DIM * sizeof(u16);
  u16* WhB = (u16*)(ws + off);   off += (size_t)PROJ * MEM_DIM * sizeof(u16);
  float* bcat = (float*)(ws + off);

  // 1) pack weights/biases, convert inputs to bf16
  treelstm_pack_kernel<<<PROJ * IN_DIM / 256, 256, 0, stream>>>(
      Wioux, Wfx, Wiouh, Wfh, bioux, bfx, WxB, WhB, bcat);
  long total = (long)N * IN_DIM;
  treelstm_xbf_kernel<<<(int)((total / 4 + 255) / 256), 256, 0, stream>>>(
      inputs, xbf, total);

  // 2) bottom-up level sweep (leaves first)
  for (int d = depth - 1; d >= 0; --d) {
    int s = (1 << d) - 1;
    int e = (1 << (d + 1)) - 1;
    int L = 1 << d;
    int nblocks = (L + 31) / 32;
    treelstm_level_kernel<<<nblocks, 256, 0, stream>>>(
        xbf, hbf, WxB, WhB, bcat, fb, h_all, c_all, s, e,
        (d != depth - 1) ? 1 : 0);
  }

  // 3) root outputs
  treelstm_root_kernel<<<1, 256, 0, stream>>>(out, h_all, c_all);
}
